// HyperbolicJTNNDecoder_11656541241778
// MI455X (gfx1250) — matrix operand order repaired and checked by reading, not compiled, not run on hardware
//
#include <hip/hip_runtime.h>
#include <stdint.h>

// ---------------- problem constants ----------------
#define T_  64
#define M_  256
#define K_  6
#define B_  256
#define H_  256
#define L_  128
#define V_  780
#define E_  (T_ * M_)          // 16384
#define VP_ 784                // V padded to 16
#define ROWS_ (E_ + B_)        // 16640
#define KX_ 384                // padded inner dim for score gemm (383 -> 384)

// ---------------- bf16 helpers ----------------
typedef __bf16 bf16_t;
typedef bf16_t v16bf __attribute__((ext_vector_type(16)));
typedef float  v8f   __attribute__((ext_vector_type(8)));

__device__ __forceinline__ unsigned short f2bf(float x) {
    union { float f; uint32_t u; } v; v.f = x;
    uint32_t r = (v.u + 0x7FFFu + ((v.u >> 16) & 1u)) >> 16;
    return (unsigned short)r;
}

__device__ __forceinline__ float waveSum(float v) {
    #pragma unroll
    for (int o = 16; o > 0; o >>= 1) v += __shfl_xor(v, o, 32);
    return v;
}
__device__ __forceinline__ float waveMax(float v) {
    #pragma unroll
    for (int o = 16; o > 0; o >>= 1) v = fmaxf(v, __shfl_xor(v, o, 32));
    return v;
}

// build one 16-half fragment pair element (A layout) -------------------------
// A (16x32 bf16): lane = hi*16+lo; m = lo; VGPR v<4: K = 2v + 8*hi; v>=4: K = 16+2(v-4)+8*hi
// B (32x16 bf16): lane = hi*16+lo; n = lo; VGPR v<4: K = 2v + 8*hi; v>=4: K = 16+2(v-4)+8*hi
// C (16x16 f32) : row = v + 8*hi, col = lo

// ---------------- init / conversion kernels ----------------
__global__ void k_zero(float* acc) {
    if (threadIdx.x < 8) acc[threadIdx.x] = 0.f;
}

// one block per hbuf row (E_+1 rows): [1,0,...,0]
__global__ void k_inithbuf(float* hF, unsigned short* hB) {
    size_t g = (size_t)blockIdx.x * 256 + threadIdx.x;
    float v = (threadIdx.x == 0) ? 1.f : 0.f;
    hF[g] = v;
    hB[g] = f2bf(v);
}

__global__ void k_convW0(const float* __restrict__ W0, unsigned short* __restrict__ W0b) {
    size_t g = (size_t)blockIdx.x * blockDim.x + threadIdx.x;  // 256*256
    W0b[g] = f2bf(W0[g]);
}

// W1: (256 x 511) -> (256 x 512) padded
__global__ void k_convW1(const float* __restrict__ W1, unsigned short* __restrict__ W1b) {
    size_t g = (size_t)blockIdx.x * blockDim.x + threadIdx.x;  // 256*512
    int j = (int)(g >> 9), k = (int)(g & 511);
    float v = (k < 511) ? W1[(size_t)j * 511 + k] : 0.f;
    W1b[g] = f2bf(v);
}

// gcls: (784 x 384); col0 = -wcls[v][0], cols1..382 = wcls[v][k], pads = 0
__global__ void k_convG(const float* __restrict__ wcls, unsigned short* __restrict__ G) {
    size_t g = (size_t)blockIdx.x * blockDim.x + threadIdx.x;
    if (g >= (size_t)VP_ * KX_) return;
    int v = (int)(g / KX_), k = (int)(g % KX_);
    float val = 0.f;
    if (v < V_) {
        if (k == 0)       val = -wcls[(size_t)v * 383];
        else if (k < 383) val =  wcls[(size_t)v * 383 + k];
    }
    G[g] = f2bf(val);
}

// ---------------- WMMA GEMM (global A w/ gather) + hyperbolic llinear -------
// out[i][j] = llinear( sum_k A[arow(i)][k] * Bm[j][k] + bias[j] )  (x @ W^T + b)
// 512 threads = 16 waves; wave w computes column-tile w; grid.x = rows/16.
__global__ __launch_bounds__(512) void k_wmma_llinear(
    const unsigned short* __restrict__ A, const int* __restrict__ gather, int astride,
    const unsigned short* __restrict__ Bm, int bstride,
    const float* __restrict__ bias, const float* __restrict__ sptr,
    float* __restrict__ outF, unsigned short* __restrict__ outB, int ktiles)
{
    __shared__ float lds[16 * 256];
    __shared__ float rtime[16], rscale[16];

    const int tid  = threadIdx.x;
    const int wave = tid >> 5, lane = tid & 31;
    const int hi = lane >> 4, lo = lane & 15;
    const int rb = blockIdx.x;

    int arow = rb * 16 + lo;
    if (gather) arow = gather[arow];
    const unsigned short* pa = A + (size_t)arow * astride;

    const int j = wave * 16 + lo;                 // output column
    const unsigned short* pb = Bm + (size_t)j * bstride;

    v8f c = {};
    for (int kt = 0; kt < ktiles; ++kt) {
        const int kbase = kt * 32;
        v16bf a, b;
        #pragma unroll
        for (int v = 0; v < 8; ++v) {
            int ka = kbase + ((v < 4) ? 2 * v : 16 + 2 * (v - 4)) + 8 * hi;
            union { uint32_t u; bf16_t h[2]; } ua;
            ua.u = *(const uint32_t*)(pa + ka);
            a[2 * v] = ua.h[0]; a[2 * v + 1] = ua.h[1];
            int kb = kbase + ((v < 4) ? 2 * v : 16 + 2 * (v - 4)) + 8 * hi;
            union { uint32_t u; bf16_t h[2]; } ub;
            ub.u = *(const uint32_t*)(pb + kb);
            b[2 * v] = ub.h[0]; b[2 * v + 1] = ub.h[1];
        }
        c = __builtin_amdgcn_wmma_f32_16x16x32_bf16(false, a, false, b, (short)0, c, false, false);
    }

    const float bj = bias[j];
    #pragma unroll
    for (int v = 0; v < 8; ++v) {
        int r = v + 8 * hi;                       // local row 0..15
        lds[r * 256 + j] = c[v] + bj;
    }
    __syncthreads();

    // row-wise normalization: wave w handles row w
    {
        const float sexp = expf(*sptr);
        int r = wave;
        float s = 0.f;
        for (int col = lane; col < 256; col += 32) {
            float x = lds[r * 256 + col];
            if (col > 0) s += x * x;
        }
        s = waveSum(s);
        if (lane == 0) {
            float x0  = lds[r * 256];
            float tme = 1.f / (1.f + expf(-x0)) * sexp + 1.1f;
            float sq  = (tme * tme - 1.f) / fmaxf(s, 1e-8f);
            rtime[r]  = tme;
            rscale[r] = sqrtf(sq);
        }
    }
    __syncthreads();

    #pragma unroll
    for (int e = 0; e < 8; ++e) {
        int pos = tid + 512 * e;                  // 4096 elements
        int r = pos >> 8, col = pos & 255;
        float val = (col == 0) ? rtime[r] : lds[pos] * rscale[r];
        size_t go = (size_t)(rb * 16 + r) * 256 + col;
        outF[go] = val;
        if (outB) outB[go] = f2bf(val);
    }
}

// ---- fused: midpoint over K + concat(cur_x, h1m) in LDS + W1 WMMA + llinear
// grid.x = M_/16; each block handles 16 nodes. A rows built in LDS (bf16),
// WMMA A operand sourced from LDS (ds_load feeding v_wmma).
__global__ __launch_bounds__(512) void k_mid_wmma(
    const float* __restrict__ h1, const float* __restrict__ w6,
    const int* __restrict__ widt, const float* __restrict__ emb,
    const unsigned short* __restrict__ W1b,
    const float* __restrict__ bias, const float* __restrict__ sptr,
    float* __restrict__ outF, unsigned short* __restrict__ outB)
{
    __shared__ unsigned short ccs[16 * 512];      // 16 concat rows (bf16), 16 KB
    __shared__ float lds[16 * 256];               // GEMM result tile, 16 KB
    __shared__ float rtime[16], rscale[16];

    const int tid  = threadIdx.x;
    const int wave = tid >> 5, lane = tid & 31;
    const int hi = lane >> 4, lo = lane & 15;
    const int rb = blockIdx.x;

    // ---- Phase 1: wave w computes midpoint + concat row for node rb*16+w
    {
        const int m = rb * 16 + wave;
        float wsum = 0.f;
        #pragma unroll
        for (int k = 0; k < K_; ++k) wsum += w6[m * K_ + k];
        wsum = fmaxf(wsum, 1e-8f);

        float ave[8];
        float sq = 0.f;
        #pragma unroll
        for (int e = 0; e < 8; ++e) {
            int hcol = lane + 32 * e;
            float a = 0.f;
            #pragma unroll
            for (int k = 0; k < K_; ++k)
                a += w6[m * K_ + k] * h1[(size_t)(m * K_ + k) * 256 + hcol];
            a /= wsum;
            ave[e] = a;
            sq += a * a;
        }
        sq = waveSum(sq);
        float a0 = __shfl(ave[0], 0, 32);         // midpoint col 0
        float rn = rsqrtf(fmaxf(-(sq - 2.f * a0 * a0), 1e-8f));

        const float* ax = emb + (size_t)widt[m] * 256;
        float h1m0  = a0 * rn;
        float curx0 = ax[0];
        float tme = sqrtf(fmaxf(curx0 * curx0 + h1m0 * h1m0 - 1.f, 1e-8f));
        #pragma unroll
        for (int e = 0; e < 8; ++e) {
            int hcol = lane + 32 * e;
            if (hcol == 0) {
                ccs[wave * 512 + 0]   = f2bf(tme);
                ccs[wave * 512 + 511] = f2bf(0.f);
            } else {
                ccs[wave * 512 + hcol]       = f2bf(ax[hcol]);      // cur_x part
                ccs[wave * 512 + 255 + hcol] = f2bf(ave[e] * rn);   // h1m part
            }
        }
    }
    __syncthreads();

    // ---- Phase 2: WMMA GEMM, A from LDS, B = W1b from global
    const unsigned short* pa = ccs + (size_t)lo * 512;
    const int j = wave * 16 + lo;
    const unsigned short* pb = W1b + (size_t)j * 512;

    v8f c = {};
    for (int kt = 0; kt < 16; ++kt) {
        const int kbase = kt * 32;
        v16bf a, b;
        #pragma unroll
        for (int v = 0; v < 8; ++v) {
            int ka = kbase + ((v < 4) ? 2 * v : 16 + 2 * (v - 4)) + 8 * hi;
            union { uint32_t u; bf16_t h[2]; } ua;
            ua.u = *(const uint32_t*)(pa + ka);
            a[2 * v] = ua.h[0]; a[2 * v + 1] = ua.h[1];
            int kb = kbase + ((v < 4) ? 2 * v : 16 + 2 * (v - 4)) + 8 * hi;
            union { uint32_t u; bf16_t h[2]; } ub;
            ub.u = *(const uint32_t*)(pb + kb);
            b[2 * v] = ub.h[0]; b[2 * v + 1] = ub.h[1];
        }
        c = __builtin_amdgcn_wmma_f32_16x16x32_bf16(false, a, false, b, (short)0, c, false, false);
    }
    __syncthreads();                               // ccs no longer needed

    const float bj = bias[j];
    #pragma unroll
    for (int v = 0; v < 8; ++v) {
        int r = v + 8 * hi;
        lds[r * 256 + j] = c[v] + bj;
    }
    __syncthreads();

    {
        const float sexp = expf(*sptr);
        int r = wave;
        float s = 0.f;
        for (int col = lane; col < 256; col += 32) {
            float x = lds[r * 256 + col];
            if (col > 0) s += x * x;
        }
        s = waveSum(s);
        if (lane == 0) {
            float x0  = lds[r * 256];
            float tme = 1.f / (1.f + expf(-x0)) * sexp + 1.1f;
            float sq  = (tme * tme - 1.f) / fmaxf(s, 1e-8f);
            rtime[r]  = tme;
            rscale[r] = sqrtf(sq);
        }
    }
    __syncthreads();

    #pragma unroll
    for (int e = 0; e < 8; ++e) {
        int pos = tid + 512 * e;
        int r = pos >> 8, col = pos & 255;
        float val = (col == 0) ? rtime[r] : lds[pos] * rscale[r];
        size_t go = (size_t)(rb * 16 + r) * 256 + col;
        outF[go] = val;
        outB[go] = f2bf(val);
    }
}

// ---------------- fused: X-row build (LDS) + score GEMM (WMMA) + CE ---------
__global__ __launch_bounds__(512) void k_scoreCE(
    const float* __restrict__ hbuf, const int* __restrict__ bidx,
    const float* __restrict__ xtv,
    const unsigned short* __restrict__ G, const float* __restrict__ wbias,
    const int* __restrict__ dir, const int* __restrict__ ptgt,
    const int* __restrict__ rwid, float* __restrict__ acc)
{
    __shared__ unsigned short xs[16 * KX_];       // 16 X rows (bf16), 12 KB
    __shared__ float sc[16 * VP_];                // 16 rows x 784 scores, ~50 KB
    const int tid  = threadIdx.x;
    const int wave = tid >> 5, lane = tid & 31;
    const int hi = lane >> 4, lo = lane & 15;
    const int rb = blockIdx.x;

    // ---- Phase 0: wave w builds X row for gi = rb*16+w
    {
        const int gi = rb * 16 + wave;
        const float* nh = nullptr;
        const float* c;
        float time;
        if (gi < E_) {
            nh = hbuf + (size_t)gi * 256;
            c  = xtv + (size_t)bidx[gi] * L_;
            float n0 = nh[0], c0 = c[0];
            time = sqrtf(fmaxf(n0 * n0 + c0 * c0 - 1.f, 1e-8f));
        } else {
            c = xtv + (size_t)(gi - E_) * L_;
            float c0 = c[0];
            time = sqrtf(fmaxf(c0 * c0, 1e-8f));  // origin time = 1
        }
        for (int col = lane; col < KX_; col += 32) {
            float v;
            if (col == 0)        v = time;
            else if (col < 256)  v = nh ? nh[col] : 0.f;
            else if (col < 383)  v = c[col - 255];
            else                 v = 0.f;
            xs[wave * KX_ + col] = f2bf(v);
        }
    }
    __syncthreads();

    // ---- Phase 1: WMMA score GEMM, A from LDS
    const unsigned short* pa = xs + (size_t)lo * KX_;
    for (int ct = wave; ct < VP_ / 16; ct += 16) {
        const int j = ct * 16 + lo;               // class index
        const unsigned short* pb = G + (size_t)j * KX_;
        v8f c = {};
        for (int kt = 0; kt < KX_ / 32; ++kt) {
            const int kbase = kt * 32;
            v16bf a, b;
            #pragma unroll
            for (int v = 0; v < 8; ++v) {
                int ka = kbase + ((v < 4) ? 2 * v : 16 + 2 * (v - 4)) + 8 * hi;
                union { uint32_t u; bf16_t h[2]; } ua;
                ua.u = *(const uint32_t*)(pa + ka);
                a[2 * v] = ua.h[0]; a[2 * v + 1] = ua.h[1];
                int kb = kbase + ((v < 4) ? 2 * v : 16 + 2 * (v - 4)) + 8 * hi;
                union { uint32_t u; bf16_t h[2]; } ub;
                ub.u = *(const uint32_t*)(pb + kb);
                b[2 * v] = ub.h[0]; b[2 * v + 1] = ub.h[1];
            }
            c = __builtin_amdgcn_wmma_f32_16x16x32_bf16(false, a, false, b, (short)0, c, false, false);
        }
        #pragma unroll
        for (int v = 0; v < 8; ++v) {
            int r = v + 8 * hi;
            float val = (j < V_) ? (2.f + 2.f * c[v] + wbias[j]) : -1e30f;
            sc[r * VP_ + j] = val;
        }
    }
    __syncthreads();

    // ---- Phase 2: wave w reduces row w (softmax CE + argmax)
    const int r  = wave;
    const int gi = rb * 16 + r;
    float mx = -3.4e38f;
    for (int col = lane; col < VP_; col += 32) mx = fmaxf(mx, sc[r * VP_ + col]);
    mx = waveMax(mx);

    float sum = 0.f, best = -3.4e38f; int bestI = 0;
    for (int col = lane; col < VP_; col += 32) {
        float v = sc[r * VP_ + col];
        sum += expf(v - mx);
        if (v > best) { best = v; bestI = col; }
    }
    sum = waveSum(sum);
    #pragma unroll
    for (int o = 16; o > 0; o >>= 1) {
        float ob = __shfl_xor(best, o, 32);
        int   oi = __shfl_xor(bestI, o, 32);
        if (ob > best || (ob == best && oi < bestI)) { best = ob; bestI = oi; }
    }

    if (lane == 0) {
        float wt; int tgt;
        if (gi < E_) { wt = (float)dir[gi]; tgt = ptgt[gi]; }
        else         { wt = 1.f;            tgt = rwid[gi - E_]; }
        float lp = sc[r * VP_ + tgt] - mx - logf(sum);
        if (wt != 0.f) {
            atomicAdd(&acc[0], wt * (-lp));
            if (bestI == tgt) atomicAdd(&acc[1], wt);
        }
        if (gi < E_) atomicAdd(&acc[2], wt);      // pmask sum
    }
}

// ---------------- stop head: midpoint + double-concat + 2-class CE ----------
__global__ __launch_bounds__(128) void k_stop(
    const float* __restrict__ hbuf,
    const int* __restrict__ noi, const float* __restrict__ now_,
    const int* __restrict__ roi, const float* __restrict__ row_,
    const int* __restrict__ wid, const int* __restrict__ rwid,
    const int* __restrict__ bidx, const float* __restrict__ xtv,
    const float* __restrict__ emb, const float* __restrict__ ucls,
    const float* __restrict__ ubias, const int* __restrict__ dir,
    float* __restrict__ acc)
{
    __shared__ float bvec[256];
    __shared__ float red[128];
    const int i = blockIdx.x, tid = threadIdx.x;

    const int* idx; const float* w; const float* a; const float* c; int tgt;
    if (i < E_) {
        idx = noi + (size_t)i * K_;  w = now_ + (size_t)i * K_;
        a = emb + (size_t)wid[i] * 256;
        c = xtv + (size_t)bidx[i] * L_;
        tgt = dir[i];
    } else {
        int jj = i - E_;
        idx = roi + (size_t)jj * K_; w = row_ + (size_t)jj * K_;
        a = emb + (size_t)rwid[jj] * 256;
        c = xtv + (size_t)jj * L_;
        tgt = 0;
    }

    float wsum = 0.f;
    #pragma unroll
    for (int k = 0; k < K_; ++k) wsum += w[k];
    wsum = fmaxf(wsum, 1e-8f);

    float sq = 0.f;
    for (int h = tid; h < 256; h += 128) {
        float ave = 0.f;
        #pragma unroll
        for (int k = 0; k < K_; ++k) ave += w[k] * hbuf[(size_t)idx[k] * 256 + h];
        ave /= wsum;
        bvec[h] = ave;
        sq += ave * ave;
    }
    red[tid] = sq;
    __syncthreads();
    for (int s = 64; s > 0; s >>= 1) { if (tid < s) red[tid] += red[tid + s]; __syncthreads(); }

    float total = red[0];
    float a0v   = bvec[0];
    float rn    = rsqrtf(fmaxf(-(total - 2.f * a0v * a0v), 1e-8f));
    __syncthreads();
    for (int h = tid; h < 256; h += 128) bvec[h] *= rn;
    __syncthreads();

    float aa0 = a[0], b0v = bvec[0], c0 = c[0];
    float t1sq = fmaxf(aa0 * aa0 + b0v * b0v - 1.f, 1e-8f);
    float x0   = sqrtf(fmaxf(t1sq + c0 * c0 - 1.f, 1e-8f));

    float p0 = 0.f, p1 = 0.f;
    for (int d = 1 + tid; d < 638; d += 128) {
        float xv;
        if (d < 256)      xv = a[d];
        else if (d < 511) xv = bvec[d - 255];
        else              xv = c[d - 510];
        p0 += xv * ucls[d];
        p1 += xv * ucls[638 + d];
    }
    if (tid == 0) { p0 -= x0 * ucls[0]; p1 -= x0 * ucls[638]; }

    red[tid] = p0; __syncthreads();
    for (int s = 64; s > 0; s >>= 1) { if (tid < s) red[tid] += red[tid + s]; __syncthreads(); }
    float s0v = red[0];
    __syncthreads();
    red[tid] = p1; __syncthreads();
    for (int s = 64; s > 0; s >>= 1) { if (tid < s) red[tid] += red[tid + s]; __syncthreads(); }
    float s1v = red[0];

    if (tid == 0) {
        float sc0 = 2.f + 2.f * s0v + ubias[0];
        float sc1 = 2.f + 2.f * s1v + ubias[1];
        float m = fmaxf(sc0, sc1);
        float lse = m + logf(expf(sc0 - m) + expf(sc1 - m));
        float st = (tgt == 0) ? sc0 : sc1;
        atomicAdd(&acc[3], lse - st);
        int am = (sc1 > sc0) ? 1 : 0;
        if (am == tgt) atomicAdd(&acc[4], 1.f);
    }
}

__global__ void k_final(const float* __restrict__ acc, float* __restrict__ out) {
    if (threadIdx.x == 0) {
        out[0] = acc[0] / (float)B_;                    // pred_loss
        out[1] = acc[3] / (float)B_;                    // stop_loss
        out[2] = acc[1] / ((float)B_ + acc[2]);         // pred_acc
        out[3] = acc[4] / (float)ROWS_;                 // stop_acc
    }
}

// ---------------- host launcher ----------------
extern "C" void kernel_launch(void* const* d_in, const int* in_sizes, int n_in,
                              void* d_out, int out_size, void* d_ws, size_t ws_size,
                              hipStream_t stream) {
    (void)in_sizes; (void)n_in; (void)out_size; (void)ws_size;

    const int*   wid   = (const int*)  d_in[0];
    const int*   nhi   = (const int*)  d_in[1];
    const float* nhw   = (const float*)d_in[2];
    const int*   noi   = (const int*)  d_in[3];
    const float* now_  = (const float*)d_in[4];
    const int*   bidx  = (const int*)  d_in[5];
    const int*   dir   = (const int*)  d_in[6];
    const int*   ptgt  = (const int*)  d_in[7];
    const int*   rwid  = (const int*)  d_in[8];
    const int*   roi   = (const int*)  d_in[9];
    const float* row_  = (const float*)d_in[10];
    const float* xtv   = (const float*)d_in[11];
    const float* emb   = (const float*)d_in[12];
    const float* W0    = (const float*)d_in[13];
    const float* b0    = (const float*)d_in[14];
    const float* s0    = (const float*)d_in[15];
    const float* W1    = (const float*)d_in[16];
    const float* b1    = (const float*)d_in[17];
    const float* s1    = (const float*)d_in[18];
    const float* wcls  = (const float*)d_in[19];
    const float* wbias = (const float*)d_in[20];
    const float* ucls  = (const float*)d_in[21];
    const float* ubias = (const float*)d_in[22];
    float* out = (float*)d_out;

    // workspace carve-up (256B aligned)
    char* base = (char*)d_ws;
    size_t off = 0;
    auto carve = [&](size_t bytes) -> char* {
        char* p = base + off;
        off = (off + bytes + 255) & ~(size_t)255;
        return p;
    };
    float*          acc   = (float*)         carve(8 * sizeof(float));
    float*          hbufF = (float*)         carve((size_t)(E_ + 1) * H_ * sizeof(float));
    unsigned short* hbufB = (unsigned short*)carve((size_t)(E_ + 1) * H_ * sizeof(unsigned short));
    float*          h1    = (float*)         carve((size_t)M_ * K_ * H_ * sizeof(float));
    unsigned short* W0b   = (unsigned short*)carve((size_t)H_ * H_ * sizeof(unsigned short));
    unsigned short* W1b   = (unsigned short*)carve((size_t)H_ * 512 * sizeof(unsigned short));
    unsigned short* G     = (unsigned short*)carve((size_t)VP_ * KX_ * sizeof(unsigned short));

    // prep
    hipLaunchKernelGGL(k_zero,     dim3(1),      dim3(32),  0, stream, acc);
    hipLaunchKernelGGL(k_inithbuf, dim3(E_ + 1), dim3(256), 0, stream, hbufF, hbufB);
    hipLaunchKernelGGL(k_convW0,   dim3(256),    dim3(256), 0, stream, W0, W0b);
    hipLaunchKernelGGL(k_convW1,   dim3(512),    dim3(256), 0, stream, W1, W1b);
    hipLaunchKernelGGL(k_convG,    dim3((VP_ * KX_ + 255) / 256), dim3(256), 0, stream, wcls, G);

    // sequential scan: 64 steps, each = W0 WMMA+llinear, fused mid+concat+W1 WMMA
    for (int t = 0; t < T_; ++t) {
        hipLaunchKernelGGL(k_wmma_llinear, dim3((M_ * K_) / 16), dim3(512), 0, stream,
                           hbufB, nhi + (size_t)t * M_ * K_, H_,
                           W0b, H_, b0, s0, h1, (unsigned short*)nullptr, H_ / 32);
        hipLaunchKernelGGL(k_mid_wmma, dim3(M_ / 16), dim3(512), 0, stream,
                           h1, nhw + (size_t)t * M_ * K_, wid + (size_t)t * M_, emb,
                           W1b, b1, s1,
                           hbufF + (size_t)t * M_ * H_, hbufB + (size_t)t * M_ * H_);
    }

    // prediction scores + CE (fused X build + WMMA GEMM + softmax/argmax)
    hipLaunchKernelGGL(k_scoreCE, dim3(ROWS_ / 16), dim3(512), 0, stream,
                       hbufF, bidx, xtv, G, wbias, dir, ptgt, rwid, acc);

    // stop head
    hipLaunchKernelGGL(k_stop, dim3(ROWS_), dim3(128), 0, stream,
                       hbufF, noi, now_, roi, row_, wid, rwid, bidx, xtv,
                       emb, ucls, ubias, dir, acc);

    hipLaunchKernelGGL(k_final, dim3(1), dim3(1), 0, stream, acc, out);
}
